// MultiHeadAttention_12189117186481
// MI455X (gfx1250) — matrix operand order repaired and checked by reading, not compiled, not run on hardware
//
#include <hip/hip_runtime.h>
#include <stdint.h>
#include <stddef.h>

// ---------------------------------------------------------------------------
// Types for CDNA5 WMMA (wave32): 16x16x32 bf16 -> f32
// ---------------------------------------------------------------------------
typedef __attribute__((ext_vector_type(16))) __bf16       v16bf;
typedef __attribute__((ext_vector_type(8)))  float        v8f;
typedef __attribute__((ext_vector_type(4)))  unsigned int v4u;

namespace {
constexpr int kB  = 8;
constexpr int kS  = 1024;
constexpr int kD  = 768;
constexpr int kH  = 12;
constexpr int kDH = 64;
constexpr int kM  = kB * kS;  // 8192 token rows
}

union Frag {
  v16bf v;
  v4u   u[2];
};

// A fragment: 16x32 bf16, row-major source, lda elements.
// ISA layout: lanes 0-15 hold row M=lane, K={0..7,16..23}; lanes 16-31 same
// rows, K={8..15,24..31}.  -> two contiguous 16B loads per lane.
__device__ inline v16bf load_a_frag(const __bf16* __restrict__ p, int lda) {
  const int lane = threadIdx.x & 31;
  const int r    = lane & 15;
  const int half = lane >> 4;
  const __bf16* row = p + (size_t)r * lda + half * 8;
  Frag f;
  f.u[0] = *(const v4u*)(row);
  f.u[1] = *(const v4u*)(row + 16);
  return f.v;
}

// B fragment: 32x16 bf16 where the K dimension is contiguous in memory per
// output column n (source is [N, K] row-major, ldb elements).
// ISA layout: lanes 0-15 (n=lane) hold K={0..7,16..23}, lanes 16-31 (n=lane-16)
// hold K={8..15,24..31}.  -> two contiguous 16B loads per lane.
__device__ inline v16bf load_b_frag(const __bf16* __restrict__ p, int ldb) {
  const int lane = threadIdx.x & 31;
  const int n    = lane & 15;
  const int kb   = (lane >> 4) * 8;
  const __bf16* q = p + (size_t)n * ldb + kb;
  Frag f;
  f.u[0] = *(const v4u*)(q);
  f.u[1] = *(const v4u*)(q + 16);
  return f.v;
}

__device__ inline v8f wmma_bf16(v16bf a, v16bf b, v8f c) {
  return __builtin_amdgcn_wmma_f32_16x16x32_bf16(
      /*neg_a=*/false, a, /*neg_b=*/false, b,
      /*c_mod=*/(short)0, c, /*reuse_a=*/false, /*reuse_b=*/false);
}

// Async global -> LDS copy (16B per lane), tracked by ASYNCcnt.
// lds_off: workgroup-relative LDS byte offset (low 32 bits of the flat
// address of a __shared__ object, per the ISA aperture mapping).
__device__ inline void async_copy_b128(uint32_t lds_off, const void* gaddr) {
  asm volatile("global_load_async_to_lds_b128 %0, %1, off"
               :
               : "v"(lds_off), "v"((unsigned long long)(uintptr_t)gaddr)
               : "memory");
}
__device__ inline void wait_async0() {
  asm volatile("s_wait_asynccnt 0" ::: "memory");
}
__device__ inline uint32_t lds_off(const void* p) {
  return (uint32_t)(uintptr_t)p;
}

// Row reductions across the 16-lane half-groups of the C-fragment layout.
__device__ inline float rmax16(float v) {
#pragma unroll
  for (int m = 1; m < 16; m <<= 1) v = fmaxf(v, __shfl_xor(v, m, 32));
  return v;
}
__device__ inline float rsum16(float v) {
#pragma unroll
  for (int m = 1; m < 16; m <<= 1) v += __shfl_xor(v, m, 32);
  return v;
}

// ---------------------------------------------------------------------------
// fp32 -> bf16 conversion
// ---------------------------------------------------------------------------
__global__ void cvt_f32_bf16(const float* __restrict__ in,
                             __bf16* __restrict__ out, int n) {
  int i = blockIdx.x * blockDim.x + threadIdx.x;
  if (i < n) out[i] = (__bf16)in[i];
}

// ---------------------------------------------------------------------------
// GEMM: out = A[M,K] * W[N,K]^T   (torch Linear convention, K contiguous both)
// Block = 256 threads = 8 waves as 2(M) x 4(N); block tile 64x256, BK=32.
// Operand staging: double-buffered async global->LDS copies (ASYNCcnt),
// fragments read from LDS (ds_load_b128), 8 WMMAs per wave per k-step.
// All 6 fragments are preloaded into distinct registers per k-step so the
// ds_loads batch up and drain with staged s_wait_dscnt instead of
// serializing each WMMA pair behind a wait 0.
// mode 0: bf16 out as [B,H,S,DH]          (Q / K heads-split)
// mode 1: bf16 out as [B,H,DH,S]          (V transposed for PV fragment loads)
// mode 2: f32  out as [M,N] + bias        (final projection)
// ---------------------------------------------------------------------------
__global__ __launch_bounds__(256) void gemm_kernel(
    const __bf16* __restrict__ A, const __bf16* __restrict__ W,
    const float* __restrict__ bias, void* __restrict__ out,
    int M, int N, int Kd, int mode) {
  __shared__ __bf16 Ab[2][64 * 32];    //  2 x 4 KB
  __shared__ __bf16 Bb[2][256 * 32];   //  2 x 16 KB

  const int tid  = threadIdx.x;
  const int wid  = tid >> 5;
  const int lane = tid & 31;
  const int half = lane >> 4;
  const int nidx = lane & 15;
  const int wm   = wid & 1;
  const int wn   = wid >> 1;
  const int rowblk = blockIdx.x * 64;
  const int colblk = blockIdx.y * 256;

  // Copy assignment: thread -> one 16B chunk of the A tile, four of B.
  const int crow   = tid >> 2;        // 0..63
  const int cchunk = (tid & 3) * 8;   // element offset within the 32-wide k

  auto issue = [&](int buf, int k0) {
    async_copy_b128(lds_off(&Ab[buf][crow * 32 + cchunk]),
                    A + (size_t)(rowblk + crow) * Kd + k0 + cchunk);
#pragma unroll
    for (int j = 0; j < 4; ++j) {
      const int r = crow + 64 * j;    // 0..255
      async_copy_b128(lds_off(&Bb[buf][r * 32 + cchunk]),
                      W + (size_t)(colblk + r) * Kd + k0 + cchunk);
    }
  };

  v8f acc[2][4];
#pragma unroll
  for (int mt = 0; mt < 2; ++mt)
#pragma unroll
    for (int nt = 0; nt < 4; ++nt)
      acc[mt][nt] = v8f{0.f, 0.f, 0.f, 0.f, 0.f, 0.f, 0.f, 0.f};

  const int nsteps = Kd / 32;
  issue(0, 0);
  for (int ks = 0; ks < nsteps; ++ks) {
    wait_async0();        // this wave's copies for buffer (ks&1) have landed
    __syncthreads();      // everyone's copies landed; prev buffer free
    if (ks + 1 < nsteps) issue((ks + 1) & 1, (ks + 1) * 32);

    const __bf16* Abuf = &Ab[ks & 1][0];
    const __bf16* Bbuf = &Bb[ks & 1][0];
    // Batch all LDS fragment loads into distinct registers first.
    const v16bf a0 = load_a_frag(Abuf + (size_t)(wm * 32) * 32, 32);
    const v16bf a1 = load_a_frag(Abuf + (size_t)(wm * 32 + 16) * 32, 32);
    v16bf bb[4];
#pragma unroll
    for (int nt = 0; nt < 4; ++nt)
      bb[nt] = load_b_frag(Bbuf + (size_t)(wn * 64 + nt * 16) * 32, 32);
#pragma unroll
    for (int nt = 0; nt < 4; ++nt) {
      acc[0][nt] = wmma_bf16(a0, bb[nt], acc[0][nt]);
      acc[1][nt] = wmma_bf16(a1, bb[nt], acc[1][nt]);
    }
  }

#pragma unroll
  for (int mt = 0; mt < 2; ++mt) {
#pragma unroll
    for (int i = 0; i < 8; ++i) {
      const int r = rowblk + wm * 32 + mt * 16 + half * 8 + i;
#pragma unroll
      for (int nt = 0; nt < 4; ++nt) {
        const int c = colblk + wn * 64 + nt * 16 + nidx;
        const float v = acc[mt][nt][i];
        if (mode == 2) {
          ((float*)out)[(size_t)r * N + c] = v + bias[c];
        } else {
          const int bi = r / kS, si = r % kS;
          const int hi = c / kDH, di = c % kDH;
          size_t off;
          if (mode == 0)
            off = ((size_t)(bi * kH + hi) * kS + si) * kDH + di;
          else
            off = ((size_t)(bi * kH + hi) * kDH + di) * kS + si;
          ((__bf16*)out)[off] = (__bf16)v;
        }
      }
    }
  }
}

// ---------------------------------------------------------------------------
// Causal flash attention.  One wave = one 16-row q tile (block = 8 waves =
// 128 q rows).  Q,K: [B,H,S,DH] bf16; Vt: [B,H,DH,S] bf16.
// Scores/softmax in C-fragment layout; P re-laid out via wave-private LDS
// tile into A-fragment layout for the PV WMMAs.  O written as [B,S,D] bf16.
// ---------------------------------------------------------------------------
__global__ __launch_bounds__(256) void attn_kernel(
    const __bf16* __restrict__ Q, const __bf16* __restrict__ K,
    const __bf16* __restrict__ Vt, __bf16* __restrict__ O) {
  __shared__ __bf16 pbuf[8 * 16 * 32];

  const int wid  = threadIdx.x >> 5;
  const int lane = threadIdx.x & 31;
  const int half = lane >> 4;
  const int nidx = lane & 15;

  const int qblocks = kS / 128;  // 8
  const int bh = blockIdx.x / qblocks;
  const int qb = blockIdx.x % qblocks;
  const int b  = bh / kH;
  const int h  = bh % kH;
  const int q0 = qb * 128 + wid * 16;

  const __bf16* Qb = Q + ((size_t)bh * kS + q0) * kDH;
  const v16bf aq0 = load_a_frag(Qb, kDH);       // d = 0..31
  const v16bf aq1 = load_a_frag(Qb + 32, kDH);  // d = 32..63

  float m[8], l[8];
  v8f o[4];
#pragma unroll
  for (int i = 0; i < 8; ++i) { m[i] = -1e30f; l[i] = 0.f; }
#pragma unroll
  for (int t = 0; t < 4; ++t)
    o[t] = v8f{0.f, 0.f, 0.f, 0.f, 0.f, 0.f, 0.f, 0.f};

  __bf16* pw = pbuf + wid * (16 * 32);
  const float scale = 0.125f;  // 1/sqrt(64)
  const v8f vz = v8f{0.f, 0.f, 0.f, 0.f, 0.f, 0.f, 0.f, 0.f};

  for (int kk = 0; kk < q0 + 16; kk += 32) {  // causal: keys <= q0+15
    const __bf16* Kb = K + ((size_t)bh * kS + kk) * kDH;
    // Batch the 4 K fragment loads, then run the 4 score WMMAs.
    v16bf bk[4];
    bk[0] = load_b_frag(Kb, kDH);
    bk[1] = load_b_frag(Kb + 32, kDH);
    bk[2] = load_b_frag(Kb + 16 * kDH, kDH);
    bk[3] = load_b_frag(Kb + 16 * kDH + 32, kDH);
    v8f s0 = vz, s1 = vz;
    s0 = wmma_bf16(aq0, bk[0], s0);
    s0 = wmma_bf16(aq1, bk[1], s0);
    s1 = wmma_bf16(aq0, bk[2], s1);
    s1 = wmma_bf16(aq1, bk[3], s1);

    // Prefetch V fragments for this k-step while softmax runs.
    const __bf16* Vb = Vt + (size_t)bh * kDH * kS + kk;
    v16bf bv[4];
#pragma unroll
    for (int t = 0; t < 4; ++t)
      bv[t] = load_b_frag(Vb + (size_t)(t * 16) * kS, kS);

#pragma unroll
    for (int i = 0; i < 8; ++i) {
      const int qrow = q0 + half * 8 + i;
      const float e0 = (kk + nidx      <= qrow) ? s0[i] * scale : -1e30f;
      const float e1 = (kk + 16 + nidx <= qrow) ? s1[i] * scale : -1e30f;
      const float rm   = rmax16(fmaxf(e0, e1));
      const float mn   = fmaxf(m[i], rm);
      const float corr = __expf(m[i] - mn);
      const float p0   = __expf(e0 - mn);
      const float p1   = __expf(e1 - mn);
      l[i] = l[i] * corr + rsum16(p0 + p1);
      m[i] = mn;
#pragma unroll
      for (int t = 0; t < 4; ++t) o[t][i] *= corr;
      // C layout -> LDS (16 rows x 32 keys, row-major)
      pw[(half * 8 + i) * 32 + nidx]      = (__bf16)p0;
      pw[(half * 8 + i) * 32 + 16 + nidx] = (__bf16)p1;
    }

    const v16bf ap = load_a_frag(pw, 32);  // P as A fragment (16x32)
#pragma unroll
    for (int t = 0; t < 4; ++t)
      o[t] = wmma_bf16(ap, bv[t], o[t]);
  }

#pragma unroll
  for (int i = 0; i < 8; ++i) {
    const float inv = 1.0f / l[i];
    const int row = q0 + half * 8 + i;
    const size_t base = (size_t)(b * kS + row) * kD + h * kDH + nidx;
#pragma unroll
    for (int t = 0; t < 4; ++t)
      O[base + t * 16] = (__bf16)(o[t][i] * inv);
  }
}

// ---------------------------------------------------------------------------
// Host launcher
// ---------------------------------------------------------------------------
extern "C" void kernel_launch(void* const* d_in, const int* in_sizes, int n_in,
                              void* d_out, int out_size, void* d_ws,
                              size_t ws_size, hipStream_t stream) {
  (void)in_sizes; (void)n_in; (void)out_size; (void)ws_size;
  const float* x  = (const float*)d_in[0];
  const float* wq = (const float*)d_in[1];
  const float* wk = (const float*)d_in[2];
  const float* wv = (const float*)d_in[3];
  const float* wo = (const float*)d_in[4];
  const float* bo = (const float*)d_in[5];

  char*  ws  = (char*)d_ws;
  size_t off = 0;
  auto take = [&](size_t bytes) -> char* {
    char* p = ws + off;
    off += (bytes + 255) & ~(size_t)255;
    return p;
  };

  const size_t xBytes = (size_t)kM * kD * sizeof(__bf16);
  const size_t wBytes = (size_t)kD * kD * sizeof(__bf16);

  __bf16* X16  = (__bf16*)take(xBytes);
  __bf16* Wq16 = (__bf16*)take(wBytes);
  __bf16* Wk16 = (__bf16*)take(wBytes);
  __bf16* Wv16 = (__bf16*)take(wBytes);
  __bf16* Wo16 = (__bf16*)take(wBytes);
  __bf16* Q16  = (__bf16*)take(xBytes);  // [B,H,S,DH]
  __bf16* K16  = (__bf16*)take(xBytes);  // [B,H,S,DH]
  __bf16* Vt16 = (__bf16*)take(xBytes);  // [B,H,DH,S]
  __bf16* O16  = (__bf16*)take(xBytes);  // [B,S,D]

  const int nX = kM * kD;   // 6,291,456
  const int nW = kD * kD;   // 589,824
  cvt_f32_bf16<<<(nX + 255) / 256, 256, 0, stream>>>(x, X16, nX);
  cvt_f32_bf16<<<(nW + 255) / 256, 256, 0, stream>>>(wq, Wq16, nW);
  cvt_f32_bf16<<<(nW + 255) / 256, 256, 0, stream>>>(wk, Wk16, nW);
  cvt_f32_bf16<<<(nW + 255) / 256, 256, 0, stream>>>(wv, Wv16, nW);
  cvt_f32_bf16<<<(nW + 255) / 256, 256, 0, stream>>>(wo, Wo16, nW);

  dim3 g(kM / 64, kD / 256);  // (128, 3)
  gemm_kernel<<<g, 256, 0, stream>>>(X16, Wq16, nullptr, Q16, kM, kD, kD, 0);
  gemm_kernel<<<g, 256, 0, stream>>>(X16, Wk16, nullptr, K16, kM, kD, kD, 0);
  gemm_kernel<<<g, 256, 0, stream>>>(X16, Wv16, nullptr, Vt16, kM, kD, kD, 1);

  attn_kernel<<<kB * kH * (kS / 128), 256, 0, stream>>>(Q16, K16, Vt16, O16);

  gemm_kernel<<<g, 256, 0, stream>>>(O16, Wo16, bo, d_out, kM, kD, kD, 2);
}
